// CrossImageAWGM_4260607557685
// MI455X (gfx1250) — matrix operand order repaired and checked by reading, not compiled, not run on hardware
//
#include <hip/hip_runtime.h>
#include <hip/hip_bf16.h>
#include <stdint.h>

// ---------------------------------------------------------------------------
// CrossImageAWGM on MI455X (gfx1250), bf16 WMMA path with double-buffered
// async global->LDS staging (ASYNCcnt deferred waits).
// B=16, C=512, H=W=32, P=1024.
// ---------------------------------------------------------------------------

typedef __attribute__((ext_vector_type(16))) __bf16 v16bf;
typedef __attribute__((ext_vector_type(8)))  __bf16 v8bf;
typedef __attribute__((ext_vector_type(8)))  float  v8f;

#define B_DIM   16
#define C_DIM   512
#define P_DIM   1024
#define CP      (C_DIM * P_DIM)          // 524288
#define LDT     40                        // LDS tile row stride in bf16 (32 + 8 pad, 80B rows, 16B aligned)
#define INV_P   (1.0f / 1024.0f)
#define SCALE_M (1.0f / (0.5f * 512.0f * 1024.0f))   // 1/(TAU*C*P)

// ---- CDNA5 async global->LDS copy (ASYNCcnt path) -------------------------
static __device__ __forceinline__ uint32_t lds_addr_of(const void* p) {
  return (uint32_t)(uintptr_t)p;   // generic->LDS: low 32 bits are the LDS offset
}
static __device__ __forceinline__ void async_b128(uint32_t lds_byte, const void* gptr) {
  asm volatile("global_load_async_to_lds_b128 %0, %1, off"
               :: "v"(lds_byte), "v"((uint64_t)(uintptr_t)gptr) : "memory");
}
template <int N>
static __device__ __forceinline__ void async_waitN() {
  asm volatile("s_wait_asynccnt %0" :: "i"(N) : "memory");
}

// ---- WMMA helpers ---------------------------------------------------------
static __device__ __forceinline__ v8f wmma_bf16(v16bf a, v16bf b, v8f c) {
  return __builtin_amdgcn_wmma_f32_16x16x32_bf16(false, a, false, b, (short)0, c,
                                                 false, false);
}

// A-matrix 16x32 bf16 fragment (ISA 7.12.2): lanes 0-15 row=lane K0-7/16-23,
// lanes 16-31 row=lane-16 K8-15/24-31. Tile row-major [row][k], stride LDT.
static __device__ __forceinline__ v16bf frag_a(const __bf16* tile, int row, int lane) {
  const __bf16* base = tile + row * LDT + ((lane & 16) ? 8 : 0);
  v8bf lo = *(const v8bf*)(base);
  v8bf hi = *(const v8bf*)(base + 16);
  return __builtin_shufflevector(lo, hi, 0,1,2,3,4,5,6,7,8,9,10,11,12,13,14,15);
}
// B-matrix 32x16 bf16 fragment: lane(0-15)=col n K0-7/16-23; lanes 16-31 K8-15/24-31.
// Tile holds B^T row-major: [n][k], stride LDT.
static __device__ __forceinline__ v16bf frag_b(const __bf16* tile, int col, int lane) {
  const __bf16* base = tile + col * LDT + ((lane & 16) ? 8 : 0);
  v8bf lo = *(const v8bf*)(base);
  v8bf hi = *(const v8bf*)(base + 16);
  return __builtin_shufflevector(lo, hi, 0,1,2,3,4,5,6,7,8,9,10,11,12,13,14,15);
}

// ---------------------------------------------------------------------------
// K0: fsmean = mean_b f_s ; kc = bf16(fsmean*w_cca_k) ; vcT = bf16(fsmean*w_cca_v)^T
//     plus zero m1 / qs_sum accumulators.
__global__ __launch_bounds__(256)
void k_prep(const float* __restrict__ fs, const float* __restrict__ wk,
            const float* __restrict__ wv, float* __restrict__ fsmean,
            __bf16* __restrict__ kc, __bf16* __restrict__ vcT,
            float* __restrict__ m1, float* __restrict__ qsum) {
  const int idx = blockIdx.x * 256 + threadIdx.x;   // 0..CP-1
  const int c = idx >> 10;
  const int p = idx & 1023;
  float s = 0.f;
#pragma unroll
  for (int b = 0; b < B_DIM; ++b) s += fs[(size_t)b * CP + idx];
  s *= (1.0f / 16.0f);
  fsmean[idx] = s;
  kc[idx] = (__bf16)(s * wk[c]);
  vcT[(size_t)p * C_DIM + c] = (__bf16)(s * wv[c]);
  if (idx < B_DIM * P_DIM) m1[idx] = 0.f;
  if (idx < B_DIM * C_DIM) qsum[idx] = 0.f;
}

// K0b: qc = bf16(f_q * w_cca_q)  (read f_q once instead of 4x f32 in GEMM1)
__global__ __launch_bounds__(256)
void k_qc(const float* __restrict__ fq, const float* __restrict__ wq,
          __bf16* __restrict__ qc) {
  const int idx = blockIdx.x * 256 + threadIdx.x;   // 0..B*CP-1
  const int c = (idx >> 10) & (C_DIM - 1);
  qc[idx] = (__bf16)(fq[idx] * wq[c]);
}

// K0c: ksum[c] = w_sca_k[c] * sum_q fsmean[c,q]
__global__ __launch_bounds__(256)
void k_ksum(const float* __restrict__ fsmean, const float* __restrict__ wsk,
            float* __restrict__ ksum) {
  __shared__ float red[256];
  const int c = blockIdx.x, tid = threadIdx.x;
  float a = 0.f;
  for (int q = tid; q < P_DIM; q += 256) a += fsmean[(size_t)c * P_DIM + q];
  red[tid] = a; __syncthreads();
  for (int s = 128; s; s >>= 1) { if (tid < s) red[tid] += red[tid + s]; __syncthreads(); }
  if (tid == 0) ksum[c] = wsk[c] * red[0];
}

// ---------------------------------------------------------------------------
// Shared GEMM core: block tile 128x128, BK=32, 8 waves each 32x64 (2x4 WMMA),
// both operands bf16 row-major-with-K-contiguous, double-buffered async LDS.
struct GemmRegs {
  v8f acc[2][4];
};

static __device__ __forceinline__ void gemm_zero(GemmRegs& g) {
#pragma unroll
  for (int i = 0; i < 2; ++i)
#pragma unroll
    for (int j = 0; j < 4; ++j)
#pragma unroll
      for (int r = 0; r < 8; ++r) g.acc[i][j][r] = 0.f;
}

static __device__ __forceinline__ void gemm_compute(GemmRegs& g, const __bf16* As,
                                                    const __bf16* Bs, int lane,
                                                    int wm, int wn) {
  const int lane15 = lane & 15;
  v16bf af[2], bfr[4];
#pragma unroll
  for (int mt = 0; mt < 2; ++mt) af[mt] = frag_a(As, wm * 32 + mt * 16 + lane15, lane);
#pragma unroll
  for (int nt = 0; nt < 4; ++nt) bfr[nt] = frag_b(Bs, wn * 64 + nt * 16 + lane15, lane);
#pragma unroll
  for (int mt = 0; mt < 2; ++mt)
#pragma unroll
    for (int nt = 0; nt < 4; ++nt)
      g.acc[mt][nt] = wmma_bf16(af[mt], bfr[nt], g.acc[mt][nt]);
}

// Stage one 128x32 bf16 tile from global (K-contiguous rows) via async b128.
// 256 threads: 2 threads/row, each issues 2x b128 (32B).
static __device__ __forceinline__ void stage_async(const __bf16* gbase, size_t gstride,
                                                   int k0, __bf16* lds_tile, int tid) {
  const int row = tid >> 1, half = tid & 1;
  const __bf16* g = gbase + (size_t)row * gstride + k0 + half * 16;
  const uint32_t l = lds_addr_of(&lds_tile[row * LDT + half * 16]);
  async_b128(l, g);
  async_b128(l + 16, g + 8);
}

// ---------------------------------------------------------------------------
// K1: S[b] = Qc[b] (C x P) * Kc^T  -> [C x C] f32 logits.
__global__ __launch_bounds__(256)
void k_gemm1(const __bf16* __restrict__ qc, const __bf16* __restrict__ kc,
             float* __restrict__ S) {
  __shared__ __attribute__((aligned(16))) __bf16 As[2][128 * LDT];
  __shared__ __attribute__((aligned(16))) __bf16 Bs[2][128 * LDT];
  const int b = blockIdx.z;
  const int cblk = blockIdx.y * 128;
  const int nblk = blockIdx.x * 128;
  const int tid = threadIdx.x;
  const int lane = tid & 31;
  const int wid = tid >> 5, wm = wid >> 1, wn = wid & 1;

  const __bf16* Ag = qc + (size_t)b * CP + (size_t)cblk * P_DIM;
  const __bf16* Bg = kc + (size_t)nblk * P_DIM;

  GemmRegs g; gemm_zero(g);

  // prologue: stage k0=0 into buffer 0
  stage_async(Ag, P_DIM, 0, As[0], tid);
  stage_async(Bg, P_DIM, 0, Bs[0], tid);

  const int NSTEP = P_DIM / 32;
  for (int step = 0; step < NSTEP; ++step) {
    const int cur = step & 1;
    if (step + 1 < NSTEP) {   // issue next stage before waiting: overlap with compute
      stage_async(Ag, P_DIM, (step + 1) * 32, As[1 - cur], tid);
      stage_async(Bg, P_DIM, (step + 1) * 32, Bs[1 - cur], tid);
      async_waitN<4>();       // in-order completion => current stage landed
    } else {
      async_waitN<0>();
    }
    __syncthreads();          // all waves' current-stage data visible
    gemm_compute(g, As[cur], Bs[cur], lane, wm, wn);
    __syncthreads();          // everyone done reading before buffer reuse
  }

  const int lane15 = lane & 15;
#pragma unroll
  for (int mt = 0; mt < 2; ++mt)
#pragma unroll
    for (int nt = 0; nt < 4; ++nt)
#pragma unroll
      for (int r = 0; r < 8; ++r) {
        const int c = cblk + wm * 32 + mt * 16 + r + ((lane & 16) ? 8 : 0);
        const int d = nblk + wn * 64 + nt * 16 + lane15;
        S[((size_t)b * C_DIM + c) * C_DIM + d] = g.acc[mt][nt][r];
      }
}

// K2: row softmax of S/P -> Ac bf16. One block per (b, c); row length 512.
__global__ __launch_bounds__(256)
void k_softmax_rows(const float* __restrict__ S, __bf16* __restrict__ Ac) {
  __shared__ float red[256];
  const int c = blockIdx.x, b = blockIdx.y, tid = threadIdx.x;
  const float* row = S + ((size_t)b * C_DIM + c) * C_DIM;
  const float x0 = row[tid] * INV_P, x1 = row[tid + 256] * INV_P;
  red[tid] = fmaxf(x0, x1); __syncthreads();
  for (int s = 128; s; s >>= 1) { if (tid < s) red[tid] = fmaxf(red[tid], red[tid + s]); __syncthreads(); }
  const float mx = red[0]; __syncthreads();
  const float e0 = __expf(x0 - mx), e1 = __expf(x1 - mx);
  red[tid] = e0 + e1; __syncthreads();
  for (int s = 128; s; s >>= 1) { if (tid < s) red[tid] += red[tid + s]; __syncthreads(); }
  const float inv = 1.0f / red[0];
  __bf16* orow = Ac + ((size_t)b * C_DIM + c) * C_DIM;
  orow[tid] = (__bf16)(e0 * inv);
  orow[tid + 256] = (__bf16)(e1 * inv);
}

// ---------------------------------------------------------------------------
// K3: F[b] = Ac[b] (C x C) * Vc (C x P), fused epilogue:
//   Fq_out = f_q + lam*F ;  qs = F*w_sca_q[c];  m1[b,p]+=qs*ksum[c];  qs_sum[b,c]+=qs.
__global__ __launch_bounds__(256)
void k_gemm2(const __bf16* __restrict__ Acb, const __bf16* __restrict__ vcT,
             const float* __restrict__ fq, const float* __restrict__ wsq,
             const float* __restrict__ ksum, const float* __restrict__ lamp,
             float* __restrict__ outF, float* __restrict__ m1,
             float* __restrict__ qsum) {
  __shared__ __attribute__((aligned(16))) __bf16 As[2][128 * LDT];
  __shared__ __attribute__((aligned(16))) __bf16 Bs[2][128 * LDT];
  __shared__ float sm1[128];
  __shared__ float sqc[128];
  const int b = blockIdx.z;
  const int cblk = blockIdx.y * 128;
  const int pblk = blockIdx.x * 128;
  const int tid = threadIdx.x;
  const int lane = tid & 31, lane15 = lane & 15;
  const int wid = tid >> 5, wm = wid >> 1, wn = wid & 1;
  if (tid < 128) { sm1[tid] = 0.f; sqc[tid] = 0.f; }

  const __bf16* Ag = Acb + ((size_t)b * C_DIM + cblk) * C_DIM;
  const __bf16* Bg = vcT + (size_t)pblk * C_DIM;

  GemmRegs g; gemm_zero(g);

  stage_async(Ag, C_DIM, 0, As[0], tid);
  stage_async(Bg, C_DIM, 0, Bs[0], tid);

  const int NSTEP = C_DIM / 32;
  for (int step = 0; step < NSTEP; ++step) {
    const int cur = step & 1;
    if (step + 1 < NSTEP) {
      stage_async(Ag, C_DIM, (step + 1) * 32, As[1 - cur], tid);
      stage_async(Bg, C_DIM, (step + 1) * 32, Bs[1 - cur], tid);
      async_waitN<4>();
    } else {
      async_waitN<0>();
    }
    __syncthreads();
    gemm_compute(g, As[cur], Bs[cur], lane, wm, wn);
    __syncthreads();
  }

  const float lam = lamp[0];
#pragma unroll
  for (int mt = 0; mt < 2; ++mt)
#pragma unroll
    for (int nt = 0; nt < 4; ++nt)
#pragma unroll
      for (int r = 0; r < 8; ++r) {
        const int c = cblk + wm * 32 + mt * 16 + r + ((lane & 16) ? 8 : 0);
        const int p = pblk + wn * 64 + nt * 16 + lane15;
        const float F = g.acc[mt][nt][r];
        const size_t oi = (size_t)b * CP + (size_t)c * P_DIM + p;
        outF[oi] = fq[oi] + lam * F;
        const float qs = F * wsq[c];
        atomicAdd(&sm1[p - pblk], qs * ksum[c]);
        atomicAdd(&sqc[c - cblk], qs);
      }
  __syncthreads();
  if (tid < 128) {
    atomicAdd(&m1[b * P_DIM + pblk + tid], sm1[tid]);
    atomicAdd(&qsum[b * C_DIM + cblk + tid], sqc[tid]);
  }
}

// K4: m2[b,q] = sum_c qs_sum[b,c]*w_sca_k[c]*fsmean[c,q]   (unscaled)
__global__ __launch_bounds__(256)
void k_m2(const float* __restrict__ qsum, const float* __restrict__ fsmean,
          const float* __restrict__ wsk, float* __restrict__ m2) {
  __shared__ float sk[C_DIM];
  const int b = blockIdx.y, tid = threadIdx.x;
  for (int c = tid; c < C_DIM; c += 256) sk[c] = qsum[b * C_DIM + c] * wsk[c];
  __syncthreads();
  const int q = blockIdx.x * 256 + tid;
  float acc = 0.f;
  for (int c = 0; c < C_DIM; ++c) acc += sk[c] * fsmean[(size_t)c * P_DIM + q];
  m2[b * P_DIM + q] = acc;
}

// K5: alpha[b] = softmax(SCALE_M*m1[b]); beta[b] = softmax(SCALE_M*m2[b])
static __device__ void softmax1024_block(const float* in, float* out, float* red) {
  const int tid = threadIdx.x;
  float x[4];
  float mx = -3.0e38f;
#pragma unroll
  for (int i = 0; i < 4; ++i) { x[i] = in[tid + i * 256] * SCALE_M; mx = fmaxf(mx, x[i]); }
  red[tid] = mx; __syncthreads();
  for (int s = 128; s; s >>= 1) { if (tid < s) red[tid] = fmaxf(red[tid], red[tid + s]); __syncthreads(); }
  mx = red[0]; __syncthreads();
  float e[4], sum = 0.f;
#pragma unroll
  for (int i = 0; i < 4; ++i) { e[i] = __expf(x[i] - mx); sum += e[i]; }
  red[tid] = sum; __syncthreads();
  for (int s = 128; s; s >>= 1) { if (tid < s) red[tid] += red[tid + s]; __syncthreads(); }
  const float inv = 1.0f / red[0]; __syncthreads();
#pragma unroll
  for (int i = 0; i < 4; ++i) out[tid + i * 256] = e[i] * inv;
}

__global__ __launch_bounds__(256)
void k_alpha_beta(const float* __restrict__ m1, const float* __restrict__ m2,
                  float* __restrict__ alpha_out, float* __restrict__ beta_ws) {
  __shared__ float red[256];
  const int b = blockIdx.x;
  softmax1024_block(m1 + b * P_DIM, alpha_out + b * P_DIM, red);
  __syncthreads();
  softmax1024_block(m2 + b * P_DIM, beta_ws + b * P_DIM, red);
}

// K6: beta_mean broadcast to all 16 rows of the output section
__global__ __launch_bounds__(256)
void k_beta_mean(const float* __restrict__ beta, float* __restrict__ out) {
  const int q = blockIdx.x * 256 + threadIdx.x;
  float s = 0.f;
#pragma unroll
  for (int b = 0; b < B_DIM; ++b) s += beta[b * P_DIM + q];
  s *= (1.0f / 16.0f);
#pragma unroll
  for (int b = 0; b < B_DIM; ++b) out[b * P_DIM + q] = s;
}

// ---------------------------------------------------------------------------
extern "C" void kernel_launch(void* const* d_in, const int* in_sizes, int n_in,
                              void* d_out, int out_size, void* d_ws, size_t ws_size,
                              hipStream_t stream) {
  (void)in_sizes; (void)n_in; (void)out_size; (void)ws_size;
  const float* f_q    = (const float*)d_in[0];
  const float* f_s    = (const float*)d_in[1];
  const float* w_ccaq = (const float*)d_in[2];
  const float* w_ccak = (const float*)d_in[3];
  const float* w_ccav = (const float*)d_in[4];
  const float* w_scaq = (const float*)d_in[5];
  const float* w_scak = (const float*)d_in[6];
  const float* lam    = (const float*)d_in[7];

  // workspace layout (bytes); qc (16MB) is dead after GEMM1 and is aliased by Ac.
  char* w = (char*)d_ws;
  float*  fsmean = (float*)(w + 0);                       // 2 MB
  __bf16* kc     = (__bf16*)(w + 2097152);                // 1 MB
  __bf16* vcT    = (__bf16*)(w + 3145728);                // 1 MB
  float*  m1     = (float*)(w + 4194304);                 // 64 KB
  float*  qsum   = (float*)(w + 4259840);                 // 32 KB
  float*  m2     = (float*)(w + 4292608);                 // 64 KB
  float*  beta   = (float*)(w + 4358144);                 // 64 KB
  float*  ksum   = (float*)(w + 4423680);                 // 2 KB
  float*  S      = (float*)(w + 8388608);                 // 16 MB
  __bf16* qc     = (__bf16*)(w + 25165824);               // 16 MB (GEMM1 A operand)
  __bf16* Acb    = (__bf16*)(w + 25165824);               // 8 MB (aliases qc; live after GEMM1)

  float* outF      = (float*)d_out;                       // [16,512,32,32]
  float* alpha_out = (float*)d_out + 8388608;             // [16,1024]
  float* betam_out = (float*)d_out + 8404992;             // [16,1024]

  k_prep<<<dim3(CP / 256), 256, 0, stream>>>(f_s, w_ccak, w_ccav, fsmean, kc, vcT, m1, qsum);
  k_qc<<<dim3(B_DIM * CP / 256), 256, 0, stream>>>(f_q, w_ccaq, qc);
  k_ksum<<<dim3(C_DIM), 256, 0, stream>>>(fsmean, w_scak, ksum);
  k_gemm1<<<dim3(4, 4, B_DIM), 256, 0, stream>>>(qc, kc, S);
  k_softmax_rows<<<dim3(C_DIM, B_DIM), 256, 0, stream>>>(S, Acb);
  k_gemm2<<<dim3(8, 4, B_DIM), 256, 0, stream>>>(Acb, vcT, f_q, w_scaq, ksum, lam,
                                                 outF, m1, qsum);
  k_m2<<<dim3(4, B_DIM), 256, 0, stream>>>(qsum, fsmean, w_scak, m2);
  k_alpha_beta<<<dim3(B_DIM), 256, 0, stream>>>(m1, m2, alpha_out, beta);
  k_beta_mean<<<dim3(4), 256, 0, stream>>>(beta, betam_out);
}
